// MultiHeadAttention_31241592111763
// MI455X (gfx1250) — matrix operand order repaired and checked by reading, not compiled, not run on hardware
//
#include <hip/hip_runtime.h>

typedef __attribute__((ext_vector_type(16))) __bf16 v16bf;
typedef __attribute__((ext_vector_type(8)))  __bf16 v8bf;
typedef __attribute__((ext_vector_type(8)))  float  v8f;
typedef __attribute__((ext_vector_type(4)))  unsigned int u32x4;
typedef __attribute__((ext_vector_type(8)))  int   i32x8;
typedef __attribute__((ext_vector_type(4)))  int   i32x4;

#define NB     4
#define SEQ    2048
#define DMODEL 2048
#define NH     16
#define HDIM   128

// ---------------------------------------------------------------------------
// WMMA helpers (CDNA5 16x16x32 bf16, f32 accumulate)
// ---------------------------------------------------------------------------
static __device__ __forceinline__ v8f wmma_bf16(v16bf a, v16bf b, v8f c) {
  // 8 args: (neg_a, A, neg_b, B, c_mod, C, reuse_a, reuse_b)
  return __builtin_amdgcn_wmma_f32_16x16x32_bf16(false, a, false, b, (short)0, c,
                                                 false, false);
}

// A-matrix fragment (16x32, row-major source, row stride ld).
// Per ISA 7.12.2 (16-bit A 16x32): lane m=l&15;
//   half 0 (lanes 0-15):  elems 0-7 <- K=0..7,   elems 8-15 <- K=16..23
//   half 1 (lanes 16-31): elems 0-7 <- K=8..15,  elems 8-15 <- K=24..31
static __device__ __forceinline__ v16bf load_a_frag(const __bf16* base, int ld,
                                                    int row0, int k0) {
  const int l   = threadIdx.x & 31;
  const int row = row0 + (l & 15);
  const int kb  = k0 + ((l >> 4) << 3);
  const __bf16* p = base + (long)row * ld + kb;
  v8bf lo = *(const v8bf*)(p);
  v8bf hi = *(const v8bf*)(p + 16);
  return __builtin_shufflevector(lo, hi, 0, 1, 2, 3, 4, 5, 6, 7,
                                 8, 9, 10, 11, 12, 13, 14, 15);
}

// B-matrix fragment (32x16). Source: column n of B is a contiguous run of K
// values (a row of W (N,K), or a key/hd row in LDS).
// Same K order as A: half 0 holds K=0..7,16..23; half 1 K=8..15,24..31.
static __device__ __forceinline__ v16bf load_b_frag(const __bf16* base, int ld,
                                                    int col0, int k0) {
  const int l   = threadIdx.x & 31;
  const int col = col0 + (l & 15);
  const int kb  = k0 + ((l >> 4) << 3);
  const __bf16* p = base + (long)col * ld + kb;
  v8bf lo = *(const v8bf*)(p);
  v8bf hi = *(const v8bf*)(p + 16);
  return __builtin_shufflevector(lo, hi, 0, 1, 2, 3, 4, 5, 6, 7,
                                 8, 9, 10, 11, 12, 13, 14, 15);
}

// C/D 16x16 f32 layout: element (M = r + 8*(l>>4), N = l&15) in VGPR r.
static __device__ __forceinline__ void store_tile(v8f c, void* Cout,
                                                  const float* bias, long N,
                                                  int row0, int col0,
                                                  int out_bf16) {
  const int l    = threadIdx.x & 31;
  const int half = l >> 4;
  const int cc   = l & 15;
#pragma unroll
  for (int r = 0; r < 8; ++r) {
    long row = row0 + r + half * 8;
    long col = col0 + cc;
    float v  = c[r];
    if (out_bf16) {
      ((__bf16*)Cout)[row * N + col] = (__bf16)v;
    } else {
      ((float*)Cout)[row * N + col] = v + (bias ? bias[col] : 0.0f);
    }
  }
}

// ---------------------------------------------------------------------------
// Tensor Data Mover: 2D tile load Global -> LDS (ISA 08_async_tensor, D#
// groups packed per §8.3/8.4; count=1, type=2 "image", data_size=2 bytes,
// no padding / iteration / gather / multicast). Issued once per wave; the
// issuing wave waits on TENSORcnt, block barrier publishes LDS to all waves.
// Toolchain uses the 6-arg builtin: (g0, g1, g2, g3, g4, cpol).
// ---------------------------------------------------------------------------
static __device__ __forceinline__ void tdm_load_tile_2d(
    unsigned lds_byte_addr, const void* global_tile_start,
    unsigned tile_d0_elems, unsigned tile_d1_rows,
    unsigned tensor_d0_elems, unsigned tensor_d1_rows,
    unsigned long row_stride_elems) {
  unsigned long ga = (unsigned long)global_tile_start;
  u32x4 g0;
  g0[0] = 1u;                                   // count=1, user descriptor
  g0[1] = lds_byte_addr;                        // bits 63:32 lds_addr
  g0[2] = (unsigned)(ga & 0xFFFFFFFFu);         // global_addr[31:0]
  g0[3] = (unsigned)((ga >> 32) & 0x01FFFFFFu)  // global_addr[56:32]
          | 0x80000000u;                        // type=2 (bits 127:126)
  i32x8 g1;
  g1[0] = (int)(1u << 16);                      // data_size=1 (2B), mask=0
  g1[1] = (int)((tensor_d0_elems & 0xFFFFu) << 16);
  g1[2] = (int)(((tensor_d0_elems >> 16) & 0xFFFFu) |
                ((tensor_d1_rows & 0xFFFFu) << 16));
  g1[3] = (int)(((tensor_d1_rows >> 16) & 0xFFFFu) |
                ((tile_d0_elems & 0xFFFFu) << 16));
  g1[4] = (int)(tile_d1_rows & 0xFFFFu);        // tile_dim1; tile_dim2=0
  g1[5] = (int)(row_stride_elems & 0xFFFFFFFFu);
  g1[6] = (int)((row_stride_elems >> 32) & 0xFFFFu);  // dim1_stride=0
  g1[7] = 0;
  i32x4 z4 = {0, 0, 0, 0};                      // groups 2/3 unused (2D tile)
  i32x8 z8 = {0, 0, 0, 0, 0, 0, 0, 0};
  __builtin_amdgcn_tensor_load_to_lds(g0, g1, z4, z4, z8, 0);
}

// ---------------------------------------------------------------------------
// f32 -> bf16 conversion
// ---------------------------------------------------------------------------
__global__ void f2bf_kernel(const float* __restrict__ in, __bf16* __restrict__ out,
                            long n) {
  long i = (long)blockIdx.x * blockDim.x + threadIdx.x;
  if (i < n) out[i] = (__bf16)in[i];
}

// ---------------------------------------------------------------------------
// GEMM: C(M,N) = A(M,K) @ W(N,K)^T  [+ bias], A/W bf16, C bf16 or f32
// One wave computes a 64x64 tile (4x4 WMMA grid): 16 WMMAs per 16 b128 loads
// per K-step (1.0 loads/WMMA vs 2.0 at 32x32), doubling per-wave arithmetic
// intensity. 256-thread block = 8 waves.
// ---------------------------------------------------------------------------
__global__ __launch_bounds__(256)
void gemm_xwt_kernel(const __bf16* __restrict__ A, const __bf16* __restrict__ W,
                     void* __restrict__ Cout, const float* __restrict__ bias,
                     int M, int N, int K, int out_bf16) {
  const int wave   = blockIdx.x * (blockDim.x >> 5) + (threadIdx.x >> 5);
  const int tilesN = N >> 6;
  const int tm     = wave / tilesN;
  const int tn     = wave % tilesN;
  if (tm >= (M >> 6)) return;
  const int row0 = tm << 6;
  const int col0 = tn << 6;

  v8f c[4][4] = {};
  for (int k = 0; k < K; k += 32) {
    v16bf a[4], bfr[4];
#pragma unroll
    for (int mi = 0; mi < 4; ++mi) a[mi] = load_a_frag(A, K, row0 + 16 * mi, k);
#pragma unroll
    for (int ni = 0; ni < 4; ++ni) bfr[ni] = load_b_frag(W, K, col0 + 16 * ni, k);
#pragma unroll
    for (int mi = 0; mi < 4; ++mi)
#pragma unroll
      for (int ni = 0; ni < 4; ++ni)
        c[mi][ni] = wmma_bf16(a[mi], bfr[ni], c[mi][ni]);
  }
#pragma unroll
  for (int mi = 0; mi < 4; ++mi)
#pragma unroll
    for (int ni = 0; ni < 4; ++ni)
      store_tile(c[mi][ni], Cout, bias, N, row0 + 16 * mi, col0 + 16 * ni,
                 out_bf16);
}

// ---------------------------------------------------------------------------
// RoPE in-place on Q and K bf16 buffers, layout (b*S+s, h*HD+hd)
// ---------------------------------------------------------------------------
__global__ void rope_kernel(__bf16* __restrict__ Q, __bf16* __restrict__ K,
                            long total_pairs) {
  long idx = (long)blockIdx.x * blockDim.x + threadIdx.x;
  if (idx >= total_pairs) return;
  const int PH = HDIM / 2;
  int  i = (int)(idx % PH);
  long t = idx / PH;
  int  h = (int)(t % NH); t /= NH;
  int  s = (int)(t % SEQ);
  int  b = (int)(t / SEQ);

  // inv_freq = base^(-2i/d); log(10000) = 9.210340372
  float inv = __expf(-((float)(2 * i) / (float)HDIM) * 9.210340371976184f);
  float ang = (float)s * inv;
  float sn, cs;
  __sincosf(ang, &sn, &cs);

  long base = ((long)b * SEQ + s) * DMODEL + h * HDIM + 2 * i;
  float q0 = (float)Q[base], q1 = (float)Q[base + 1];
  Q[base]     = (__bf16)(q0 * cs - q1 * sn);
  Q[base + 1] = (__bf16)(q0 * sn + q1 * cs);
  float k0 = (float)K[base], k1 = (float)K[base + 1];
  K[base]     = (__bf16)(k0 * cs - k1 * sn);
  K[base + 1] = (__bf16)(k0 * sn + k1 * cs);
}

// ---------------------------------------------------------------------------
// Causal flash attention, bf16 WMMA, f32 online softmax.
// Grid: (S/64, B*H); block: 128 threads = 4 waves; wave handles 16 q rows.
// K tile (32 keys x 128 hd) staged via the Tensor Data Mover (wave 0 issues,
// waits TENSORcnt, block barrier publishes). V tile transposed manually.
// ---------------------------------------------------------------------------
__global__ __launch_bounds__(128)
void flash_attn_kernel(const __bf16* __restrict__ Q, const __bf16* __restrict__ K,
                       const __bf16* __restrict__ V, __bf16* __restrict__ ctx) {
  __shared__ __align__(16) __bf16 Klds[32 * HDIM];    // 8 KB: [key][hd] (TDM)
  __shared__ __align__(16) __bf16 VTlds[HDIM * 32];   // 8 KB: [hd][key]
  __shared__ __align__(16) __bf16 Plds[4 * 16 * 32];  // 4 KB: per-wave P tile

  const int qb   = blockIdx.x;        // q block of 64 rows
  const int bh   = blockIdx.y;
  const int b    = bh / NH;
  const int h    = bh % NH;
  const int wid  = threadIdx.x >> 5;
  const int l    = threadIdx.x & 31;
  const int half = l >> 4;
  const int lc   = l & 15;

  const int m0blk = qb * 64;
  const int m0w   = m0blk + wid * 16;

  const __bf16* Qg = Q + ((long)b * SEQ + m0w) * DMODEL + h * HDIM;
  const __bf16* Kg = K + ((long)b * SEQ) * DMODEL + h * HDIM;
  const __bf16* Vg = V + ((long)b * SEQ) * DMODEL + h * HDIM;

  const unsigned klds_addr = (unsigned)(unsigned long)(const void*)&Klds[0];

  // preload Q fragments for the 4 HD chunks of 32
  v16bf qf[4];
#pragma unroll
  for (int c = 0; c < 4; ++c) qf[c] = load_a_frag(Qg, DMODEL, 0, c * 32);

  v8f o[8] = {};               // 16 q rows x 128 hd, C layout per 16-col tile
  float row_m[8], row_l[8];
#pragma unroll
  for (int r = 0; r < 8; ++r) { row_m[r] = -3.0e38f; row_l[r] = 0.0f; }

  const float scale  = 0.08838834764831845f;  // 1/sqrt(128)
  const int   ntiles = (m0blk + 64) >> 5;

  for (int kt = 0; kt < ntiles; ++kt) {
    const int n0 = kt << 5;
    __syncthreads();
    // --- TDM: DMA K tile (32 rows x 128 hd, row stride DMODEL) into LDS ---
    if (wid == 0) {
      tdm_load_tile_2d(klds_addr, Kg + (long)n0 * DMODEL,
                       /*tile_d0=*/HDIM, /*tile_d1=*/32,
                       /*tensor_d0=*/HDIM, /*tensor_d1=*/32,
                       /*row_stride_elems=*/DMODEL);
    }
    // --- manual transposed staging of V tile: VT[hd][kk] = V[n0+kk][hd] ---
    for (int idx = threadIdx.x; idx < 32 * HDIM; idx += blockDim.x) {
      int kk = idx >> 7;
      int hd = idx & 127;
      VTlds[hd * 32 + kk] = Vg[(long)(n0 + kk) * DMODEL + hd];
    }
    if (wid == 0) __builtin_amdgcn_s_wait_tensorcnt(0);
    __syncthreads();

    if (n0 > m0w + 15) continue;  // fully masked for this wave

    // S = Q @ K^T for two 16-wide key subtiles
    v8f s0 = {}, s1 = {};
#pragma unroll
    for (int c = 0; c < 4; ++c) {
      v16bf kf = load_b_frag(Klds, HDIM, 0, c * 32);
      s0 = wmma_bf16(qf[c], kf, s0);
    }
#pragma unroll
    for (int c = 0; c < 4; ++c) {
      v16bf kf = load_b_frag(Klds, HDIM, 16, c * 32);
      s1 = wmma_bf16(qf[c], kf, s1);
    }

    // online softmax update (rows live in (r, half); cols across 16 lanes)
    float p0[8], p1[8];
#pragma unroll
    for (int r = 0; r < 8; ++r) {
      int row = m0w + r + half * 8;
      int c0  = n0 + lc;
      int c1  = n0 + 16 + lc;
      float v0 = (c0 <= row) ? s0[r] * scale : -3.0e38f;
      float v1 = (c1 <= row) ? s1[r] * scale : -3.0e38f;
      float mx = fmaxf(v0, v1);
      mx = fmaxf(mx, __shfl_xor(mx, 1));
      mx = fmaxf(mx, __shfl_xor(mx, 2));
      mx = fmaxf(mx, __shfl_xor(mx, 4));
      mx = fmaxf(mx, __shfl_xor(mx, 8));
      float mnew  = fmaxf(row_m[r], mx);
      float alpha = __expf(row_m[r] - mnew);
      float e0 = __expf(v0 - mnew);
      float e1 = __expf(v1 - mnew);
      float sum = e0 + e1;
      sum += __shfl_xor(sum, 1);
      sum += __shfl_xor(sum, 2);
      sum += __shfl_xor(sum, 4);
      sum += __shfl_xor(sum, 8);
      row_l[r] = row_l[r] * alpha + sum;
      row_m[r] = mnew;
      p0[r] = e0;
      p1[r] = e1;
#pragma unroll
      for (int t = 0; t < 8; ++t) o[t][r] *= alpha;
    }

    // P (C layout) -> LDS -> A-layout fragment
    __bf16* Pw = &Plds[wid * 16 * 32];
#pragma unroll
    for (int r = 0; r < 8; ++r) {
      int row = r + half * 8;
      Pw[row * 32 + lc]      = (__bf16)p0[r];
      Pw[row * 32 + 16 + lc] = (__bf16)p1[r];
    }
    asm volatile("s_wait_dscnt 0" ::: "memory");
    v16bf pf = load_a_frag(Pw, 32, 0, 0);

    // O += P @ V over 8 hd tiles of 16
#pragma unroll
    for (int t = 0; t < 8; ++t) {
      v16bf vf = load_b_frag(VTlds, 32, t * 16, 0);
      o[t] = wmma_bf16(pf, vf, o[t]);
    }
  }

  // epilogue: normalize and store ctx (b*S+q, h*HD+hd) as bf16
  __bf16* Cg = ctx + ((long)b * SEQ + m0w) * DMODEL + h * HDIM;
#pragma unroll
  for (int t = 0; t < 8; ++t) {
#pragma unroll
    for (int r = 0; r < 8; ++r) {
      int row = r + half * 8;
      int col = t * 16 + lc;
      Cg[(long)row * DMODEL + col] = (__bf16)(o[t][r] / row_l[r]);
    }
  }
}

// ---------------------------------------------------------------------------
// Orchestration
// ---------------------------------------------------------------------------
extern "C" void kernel_launch(void* const* d_in, const int* in_sizes, int n_in,
                              void* d_out, int out_size, void* d_ws, size_t ws_size,
                              hipStream_t stream) {
  const float* x  = (const float*)d_in[0];
  const float* Wq = (const float*)d_in[1];
  const float* Wk = (const float*)d_in[2];
  const float* Wv = (const float*)d_in[3];
  const float* Wo = (const float*)d_in[4];
  const float* bo = (const float*)d_in[5];
  float* out = (float*)d_out;

  const long MR = (long)NB * SEQ;          // 8192 rows
  const long xN = MR * DMODEL;             // activation elements
  const long wN = (long)DMODEL * DMODEL;   // weight elements

  char* p = (char*)d_ws;
  __bf16* xb  = (__bf16*)p; p += xN * 2;
  __bf16* wqb = (__bf16*)p; p += wN * 2;
  __bf16* wkb = (__bf16*)p; p += wN * 2;
  __bf16* wvb = (__bf16*)p; p += wN * 2;
  __bf16* wob = (__bf16*)p; p += wN * 2;
  __bf16* Qb  = (__bf16*)p; p += xN * 2;
  __bf16* Kb  = (__bf16*)p; p += xN * 2;
  __bf16* Vb  = (__bf16*)p; p += xN * 2;
  __bf16* Cb  = (__bf16*)p; p += xN * 2;

  const int thr = 256;
  f2bf_kernel<<<(int)((xN + thr - 1) / thr), thr, 0, stream>>>(x, xb, xN);
  f2bf_kernel<<<(int)((wN + thr - 1) / thr), thr, 0, stream>>>(Wq, wqb, wN);
  f2bf_kernel<<<(int)((wN + thr - 1) / thr), thr, 0, stream>>>(Wk, wkb, wN);
  f2bf_kernel<<<(int)((wN + thr - 1) / thr), thr, 0, stream>>>(Wv, wvb, wN);
  f2bf_kernel<<<(int)((wN + thr - 1) / thr), thr, 0, stream>>>(Wo, wob, wN);

  const int tiles  = (int)((MR / 64) * (DMODEL / 64));  // 4096 waves
  const int blocks = tiles / 8;                          // 8 waves/block
  gemm_xwt_kernel<<<blocks, 256, 0, stream>>>(xb, wqb, Qb, nullptr,
                                              (int)MR, DMODEL, DMODEL, 1);
  gemm_xwt_kernel<<<blocks, 256, 0, stream>>>(xb, wkb, Kb, nullptr,
                                              (int)MR, DMODEL, DMODEL, 1);
  gemm_xwt_kernel<<<blocks, 256, 0, stream>>>(xb, wvb, Vb, nullptr,
                                              (int)MR, DMODEL, DMODEL, 1);

  const long pairs = (long)NB * SEQ * NH * (HDIM / 2);
  rope_kernel<<<(int)((pairs + 255) / 256), 256, 0, stream>>>(Qb, Kb, pairs);

  dim3 ag(SEQ / 64, NB * NH);
  flash_attn_kernel<<<ag, 128, 0, stream>>>(Qb, Kb, Vb, Cb);

  gemm_xwt_kernel<<<blocks, 256, 0, stream>>>(Cb, wob, out, bo,
                                              (int)MR, DMODEL, DMODEL, 0);
}
